// TimesBlock_83468394430476
// MI455X (gfx1250) — matrix operand order repaired and checked by reading, not compiled, not run on hardware
//
#include <hip/hip_runtime.h>
#include <hip/hip_bf16.h>
#include <math.h>

#define NFREQ 5
#define TLEN 1024
#define FDIM 64

typedef __attribute__((ext_vector_type(16))) _Float16 v16h;
typedef __attribute__((ext_vector_type(8)))  float    v8f;

// ---------------------------------------------------------------------------
// Kernel 1: per-(b,h) DFT amplitude at the 5 selected frequencies + softmax.
// One block per slice (256 blocks), 256 threads: thread = (chunk 0..3, f 0..63).
// Writes pw[slice][5] (float) into workspace.
// ---------------------------------------------------------------------------
__global__ __launch_bounds__(256) void tb_dft_softmax_kernel(
    const float* __restrict__ x, float* __restrict__ pw)
{
    __shared__ float s_re[NFREQ][4][FDIM];
    __shared__ float s_im[NFREQ][4][FDIM];
    __shared__ float s_mag[NFREQ][FDIM];

    const int slice = blockIdx.x;          // 0..255  (b*16 + h)
    const int tid   = threadIdx.x;
    const int f     = tid & 63;
    const int chunk = tid >> 6;            // 0..3, each covers 256 time steps

    const float* xs = x + (size_t)slice * (TLEN * FDIM);
    const int FREQS[NFREQ] = {1, 103, 205, 307, 409};
    const float TWO_PI_OVER_T = 6.28318530717958647692f / (float)TLEN;

    float re[NFREQ], im[NFREQ];
#pragma unroll
    for (int j = 0; j < NFREQ; ++j) { re[j] = 0.0f; im[j] = 0.0f; }

    for (int tt = chunk * 256; tt < chunk * 256 + 256; ++tt) {
        const float v = xs[tt * FDIM + f];
#pragma unroll
        for (int j = 0; j < NFREQ; ++j) {
            // exact modular reduction of k*t mod T keeps angles in [-2pi, 0]
            const int kt = (FREQS[j] * tt) & (TLEN - 1);
            float s, c;
            __sincosf(-TWO_PI_OVER_T * (float)kt, &s, &c);
            re[j] = fmaf(v, c, re[j]);
            im[j] = fmaf(v, s, im[j]);
        }
    }
#pragma unroll
    for (int j = 0; j < NFREQ; ++j) {
        s_re[j][chunk][f] = re[j];
        s_im[j][chunk][f] = im[j];
    }
    __syncthreads();

    if (tid < FDIM) {
#pragma unroll
        for (int j = 0; j < NFREQ; ++j) {
            const float rr = s_re[j][0][f] + s_re[j][1][f] + s_re[j][2][f] + s_re[j][3][f];
            const float ii = s_im[j][0][f] + s_im[j][1][f] + s_im[j][2][f] + s_im[j][3][f];
            s_mag[j][f] = sqrtf(rr * rr + ii * ii);
        }
    }
    __syncthreads();

    if (tid == 0) {
        float amp[NFREQ], mx = -1e30f;
#pragma unroll
        for (int j = 0; j < NFREQ; ++j) {
            float s = 0.0f;
            for (int f2 = 0; f2 < FDIM; ++f2) s += s_mag[j][f2];
            amp[j] = s * (1.0f / (float)FDIM);
            mx = fmaxf(mx, amp[j]);
        }
        float e[NFREQ], den = 0.0f;
#pragma unroll
        for (int j = 0; j < NFREQ; ++j) { e[j] = __expf(amp[j] - mx); den += e[j]; }
#pragma unroll
        for (int j = 0; j < NFREQ; ++j) pw[slice * NFREQ + j] = e[j] / den;
    }
}

// ---------------------------------------------------------------------------
// Kernel 2: implicit-GEMM 3x3 conv (per-branch p x q grid) on WMMA f16->f32,
// ReLU+bias, softmax-weighted blend over 5 branches, + residual.
// Grid: (16 t-tiles of 64, 256 slices). Block: 128 threads = 4 waves.
// Each wave owns a 16-position strip x 64 output channels.
// ---------------------------------------------------------------------------
__global__ __launch_bounds__(128) void tb_conv_wmma_kernel(
    const float* __restrict__ x, const float* __restrict__ Wg,
    const float* __restrict__ bias, const float* __restrict__ pw,
    float* __restrict__ out)
{
    // Weights as f16, transposed to [tap][fo][fi] so B-fragments are 16
    // contiguous halves per lane (two ds_load_b128).
    __shared__ __attribute__((aligned(32))) _Float16 ldsW[9 * FDIM * FDIM]; // 72 KB

    const int tid = threadIdx.x;
    for (int idx = tid; idx < 9 * FDIM * FDIM; idx += 128) {
        const int tap = idx >> 12;
        const int fi  = (idx >> 6) & 63;
        const int fo  = idx & 63;
        ldsW[(tap << 12) + (fo << 6) + fi] = (_Float16)Wg[idx]; // Wg[tap][fi][fo]
    }
    __syncthreads();

    const int lane  = tid & 31;
    const int wave  = tid >> 5;
    const int g     = lane >> 4;   // lane group (ISA wave32 fragment layouts)
    const int n16   = lane & 15;
    const int slice = blockIdx.y;
    const int m     = blockIdx.x * 64 + wave * 16 + n16; // this lane's A row = t pos

    const float* xs = x + (size_t)slice * (TLEN * FDIM);

    float bv[4];
#pragma unroll
    for (int nt = 0; nt < 4; ++nt) bv[nt] = bias[nt * 16 + n16];

    const int PERIOD[NFREQ] = {2, 104, 206, 308, 410};
    const int QDIM[NFREQ]   = {512, 10, 5, 4, 3};

    v8f sum[4] = {};

#pragma unroll
    for (int j = 0; j < NFREQ; ++j) {
        const int p = PERIOD[j];
        const int q = QDIM[j];
        const float pwj = pw[slice * NFREQ + j];
        const int r = m / q;          // q is a constant after unroll -> mul/shift
        const int c = m - r * q;

        v8f acc[4] = {};

#pragma unroll
        for (int tap = 0; tap < 9; ++tap) {
            const int dr = tap / 3 - 1;
            const int dc = tap % 3 - 1;
            const int rr = r + dr;
            const int cc = c + dc;
            const int t2 = rr * q + cc;
            const bool valid = (rr >= 0) && (rr < p) && (cc >= 0) && (cc < q) &&
                               (t2 < TLEN); // t2 in [T, p*q) is the zero-pad tail
            const float* abase = xs + (size_t)(valid ? t2 : 0) * FDIM;

#pragma unroll
            for (int kc = 0; kc < 2; ++kc) {
                // A fragment, contiguous k order (same as B below):
                // lane group g: halves 0..15 -> K = 16g+0..15
                v16h a;
                const int fi0 = kc * 32 + 16 * g;
                if (valid) {
                    const float4* ap = (const float4*)(abase + fi0);
                    const float4 u0 = ap[0];
                    const float4 u1 = ap[1];
                    const float4 u2 = ap[2]; // +8 floats
                    const float4 u3 = ap[3];
                    a[0]  = (_Float16)u0.x; a[1]  = (_Float16)u0.y;
                    a[2]  = (_Float16)u0.z; a[3]  = (_Float16)u0.w;
                    a[4]  = (_Float16)u1.x; a[5]  = (_Float16)u1.y;
                    a[6]  = (_Float16)u1.z; a[7]  = (_Float16)u1.w;
                    a[8]  = (_Float16)u2.x; a[9]  = (_Float16)u2.y;
                    a[10] = (_Float16)u2.z; a[11] = (_Float16)u2.w;
                    a[12] = (_Float16)u3.x; a[13] = (_Float16)u3.y;
                    a[14] = (_Float16)u3.z; a[15] = (_Float16)u3.w;
                } else {
#pragma unroll
                    for (int h = 0; h < 16; ++h) a[h] = (_Float16)0.0f;
                }

                const int fib = kc * 32 + 16 * g; // B 32x16: lane=N, halves=K (by group)
#pragma unroll
                for (int nt = 0; nt < 4; ++nt) {
                    const int fo = nt * 16 + n16;
                    const v16h b = *(const v16h*)(&ldsW[(tap << 12) + (fo << 6) + fib]);
                    acc[nt] = __builtin_amdgcn_wmma_f32_16x16x32_f16(
                        false, a, false, b, (short)0, acc[nt], false, false);
                }
            }
        }

        // ReLU(conv + bias) weighted by pw_j, accumulated in registers
#pragma unroll
        for (int nt = 0; nt < 4; ++nt)
#pragma unroll
            for (int v = 0; v < 8; ++v)
                sum[nt][v] = fmaf(pwj, fmaxf(acc[nt][v] + bv[nt], 0.0f), sum[nt][v]);
    }

    // D layout: lane L, vgpr v -> row = base + v + 8*(L/16), col = nt*16 + (L%16)
    const int mrow_base = blockIdx.x * 64 + wave * 16 + 8 * g;
#pragma unroll
    for (int nt = 0; nt < 4; ++nt) {
        const int fo = nt * 16 + n16;
#pragma unroll
        for (int v = 0; v < 8; ++v) {
            const size_t o = (size_t)slice * (TLEN * FDIM) +
                             (size_t)(mrow_base + v) * FDIM + fo;
            out[o] = sum[nt][v] + x[o]; // residual
        }
    }
}

// ---------------------------------------------------------------------------
extern "C" void kernel_launch(void* const* d_in, const int* in_sizes, int n_in,
                              void* d_out, int out_size, void* d_ws, size_t ws_size,
                              hipStream_t stream)
{
    (void)in_sizes; (void)n_in; (void)out_size; (void)ws_size;
    const float* x    = (const float*)d_in[0];   // (16,16,1024,64) f32
    const float* Wg   = (const float*)d_in[1];   // (1,3,3,64,64)  f32
    const float* bias = (const float*)d_in[2];   // (64,)          f32
    float* out = (float*)d_out;
    float* pw  = (float*)d_ws;                   // 256*5 floats of scratch

    // 1) spectral softmax weights
    tb_dft_softmax_kernel<<<dim3(256), dim3(256), 0, stream>>>(x, pw);

    // 2) 5-branch WMMA conv + blend + residual
    tb_conv_wmma_kernel<<<dim3(16, 256), dim3(128), 0, stream>>>(x, Wg, bias, pw, out);
}
